// conditional_transport_76046690943323
// MI455X (gfx1250) — compile-verified, matrix operand order checked by reading, not run on hardware
//
#include <hip/hip_runtime.h>

// ---------------------------------------------------------------------------
// Conditional-transport cost on MI455X (gfx1250, wave32, WMMA).
//
// cost[i,j] = |x_i|^2 + |y_j|^2 - 2 x_i.y_j     (X,Y: [8192,256] f32)
// ct = 0.5*sum(softmax(-cost,1)*cost)/N + 0.5*sum(softmax(-cost,0)*cost)/M
//
// Fully fused flash-softmax over a never-materialized 8192x8192 cost matrix.
// GEMM runs on v_wmma_f32_16x16x32_bf16 with a bf16 hi/lo split (3 WMMAs per
// K-chunk => ~f32 accuracy). The f32->bf16 split is hoisted into a one-time
// O(N*K) preprocessing pass (16 MB in ws, L2-resident) so the O(N^2) inner
// loop is pure loads + WMMA + exp instead of being conversion-VALU bound.
// Falls back to in-loop conversion if ws_size is too small.
// ---------------------------------------------------------------------------

typedef __attribute__((ext_vector_type(16))) __bf16 v16bf;
typedef __attribute__((ext_vector_type(8)))  __bf16 v8bf;
typedef __attribute__((ext_vector_type(4)))  __bf16 v4bf;
typedef __attribute__((ext_vector_type(8)))  float  v8f;

#define NROWS 8192
#define KDIM  256
#define NTILES (NROWS / 16)   // 512 column tiles
#define NWAVES 8

// ws layout: [0..1] accumulators, [16..16+8192) |x|^2, then |y|^2,
// then (optional) bf16 hi/lo copies of X and Y.
#define WS_BF16_OFF ((size_t)(16 + 2 * NROWS) * 4)          // bytes
#define MAT_BYTES   ((size_t)NROWS * KDIM * 2)              // one bf16 matrix
#define WS_NEEDED   (WS_BF16_OFF + 4 * MAT_BYTES)

// ---- helpers --------------------------------------------------------------

// Convert 8 consecutive f32 into bf16 hi/lo at vector elements [base,base+8).
__device__ __forceinline__ void cvt8(float4 f0, float4 f1,
                                     v16bf& hi, v16bf& lo, int base) {
  float v[8] = {f0.x, f0.y, f0.z, f0.w, f1.x, f1.y, f1.z, f1.w};
#pragma unroll
  for (int i = 0; i < 8; ++i) {
    __bf16 h = (__bf16)v[i];
    hi[base + i] = h;
    lo[base + i] = (__bf16)(v[i] - (float)h);
  }
}

// Load a 16-element WMMA fragment slice from a presplit bf16 row:
// elements 0..7 <- K kb..kb+7, elements 8..15 <- K kb+16..kb+23.
__device__ __forceinline__ v16bf ldfrag(const __bf16* __restrict__ row, int kb) {
  const v8bf* p = (const v8bf*)(row + kb);
  v8bf a = p[0];
  v8bf b = p[2];
  return __builtin_shufflevector(a, b, 0, 1, 2, 3, 4, 5, 6, 7,
                                 8, 9, 10, 11, 12, 13, 14, 15);
}

// ---- preprocessing --------------------------------------------------------

// Row norms of X and Y; also zero the two scalar accumulators.
__global__ void ct_norms_kernel(const float* __restrict__ X,
                                const float* __restrict__ Y,
                                float* __restrict__ ws) {
  if (blockIdx.x == 0 && threadIdx.x == 0) { ws[0] = 0.0f; ws[1] = 0.0f; }
  const int wave = threadIdx.x >> 5;
  const int lane = threadIdx.x & 31;
  int gr = blockIdx.x * 8 + wave;               // 0..16383
  const bool isX = gr < NROWS;
  int row = isX ? gr : gr - NROWS;
  const float* p = (isX ? X : Y) + (size_t)row * KDIM;
  float acc = 0.0f;
#pragma unroll
  for (int i = 0; i < 8; ++i) {
    float f = p[lane + 32 * i];
    acc += f * f;
  }
#pragma unroll
  for (int off = 1; off < 32; off <<= 1) acc += __shfl_xor(acc, off, 32);
  float* nx2 = ws + 16;
  float* ny2 = ws + 16 + NROWS;
  if (lane == 0) (isX ? nx2 : ny2)[row] = acc;
}

// One-time f32 -> bf16 hi/lo split of both matrices (4 f32 per thread).
__global__ void ct_split_kernel(const float* __restrict__ X,
                                const float* __restrict__ Y,
                                __bf16* __restrict__ hx, __bf16* __restrict__ lx,
                                __bf16* __restrict__ hy, __bf16* __restrict__ ly) {
  const size_t half = (size_t)NROWS * KDIM;
  size_t idx = ((size_t)blockIdx.x * blockDim.x + threadIdx.x) * 4;
  const float* src;
  __bf16 *dh, *dl;
  size_t off;
  if (idx < half) { src = X; dh = hx; dl = lx; off = idx; }
  else            { src = Y; dh = hy; dl = ly; off = idx - half; }
  float4 f = *(const float4*)(src + off);
  float v[4] = {f.x, f.y, f.z, f.w};
  v4bf h, l;
#pragma unroll
  for (int i = 0; i < 4; ++i) {
    __bf16 hh = (__bf16)v[i];
    h[i] = hh;
    l[i] = (__bf16)(v[i] - (float)hh);
  }
  *(v4bf*)(dh + off) = h;
  *(v4bf*)(dl + off) = l;
}

// ---- main fused GEMM + online softmax -------------------------------------

template <bool SPLIT>
__global__ void __launch_bounds__(256)
ct_main_kernel(const float* __restrict__ P, const float* __restrict__ Q,
               const __bf16* __restrict__ phi, const __bf16* __restrict__ plo,
               const __bf16* __restrict__ qhi, const __bf16* __restrict__ qlo,
               const float* __restrict__ np2, const float* __restrict__ nq2,
               float* __restrict__ accum) {
  const int wave = threadIdx.x >> 5;
  const int lane = threadIdx.x & 31;
  const int half = lane >> 4;   // selects K sub-block per WMMA layout
  const int lr   = lane & 15;   // row (A) / column (B) index within tile
  const int i0   = blockIdx.x * 16;

  // ---- Resident A block: 16 rows of P, K=256, bf16 hi/lo split ----
  v16bf a_hi[8], a_lo[8];
  if (SPLIT) {
    const __bf16* ph = phi + (size_t)(i0 + lr) * KDIM;
    const __bf16* pl = plo + (size_t)(i0 + lr) * KDIM;
#pragma unroll
    for (int kc = 0; kc < 8; ++kc) {
      a_hi[kc] = ldfrag(ph, kc * 32 + half * 8);
      a_lo[kc] = ldfrag(pl, kc * 32 + half * 8);
    }
  } else {
    const float* prow = P + (size_t)(i0 + lr) * KDIM;
#pragma unroll
    for (int kc = 0; kc < 8; ++kc) {
      const float4* p4 = (const float4*)(prow + kc * 32 + half * 8);
      cvt8(p4[0], p4[1], a_hi[kc], a_lo[kc], 0);
      cvt8(p4[4], p4[5], a_hi[kc], a_lo[kc], 8);
    }
  }

  // Row norms for the 8 rows this lane's accumulator slots map to.
  float x2r[8];
#pragma unroll
  for (int v = 0; v < 8; ++v) x2r[v] = np2[i0 + v + 8 * half];

  // Online softmax state per accumulator slot (8 rows per lane).
  float m[8], s[8], w[8];
#pragma unroll
  for (int v = 0; v < 8; ++v) { m[v] = __builtin_inff(); s[v] = 0.0f; w[v] = 0.0f; }

  // ---- Stream all column tiles of Q; wave-strided over tiles ----
  for (int t = wave; t < NTILES; t += NWAVES) {
    const int n0 = t * 16;
    const float y2c = nq2[n0 + lr];

    // Prefetch next tile's rows into cache (global_prefetch_b8).
    {
      int tn = t + NWAVES;
      if (tn < NTILES) {
        size_t roff = (size_t)(tn * 16 + lr) * KDIM;
        if (SPLIT) {
          __builtin_prefetch(qhi + roff, 0, 1);
          __builtin_prefetch(qlo + roff, 0, 1);
        } else {
          __builtin_prefetch(Q + roff, 0, 1);
          __builtin_prefetch(Q + roff + 128, 0, 1);
        }
      }
    }

    v8f c = {0.f, 0.f, 0.f, 0.f, 0.f, 0.f, 0.f, 0.f};
    if (SPLIT) {
      const __bf16* qh = qhi + (size_t)(n0 + lr) * KDIM;
      const __bf16* ql = qlo + (size_t)(n0 + lr) * KDIM;
#pragma unroll
      for (int kc = 0; kc < 8; ++kc) {
        v16bf b_hi = ldfrag(qh, kc * 32 + half * 8);
        v16bf b_lo = ldfrag(ql, kc * 32 + half * 8);
        c = __builtin_amdgcn_wmma_f32_16x16x32_bf16(false, a_hi[kc], false, b_hi,
                                                    (short)0, c, false, false);
        c = __builtin_amdgcn_wmma_f32_16x16x32_bf16(false, a_hi[kc], false, b_lo,
                                                    (short)0, c, false, false);
        c = __builtin_amdgcn_wmma_f32_16x16x32_bf16(false, a_lo[kc], false, b_hi,
                                                    (short)0, c, false, false);
      }
    } else {
      const float* qrow = Q + (size_t)(n0 + lr) * KDIM;
#pragma unroll
      for (int kc = 0; kc < 8; ++kc) {
        const float4* q4 = (const float4*)(qrow + kc * 32 + half * 8);
        v16bf b_hi, b_lo;
        cvt8(q4[0], q4[1], b_hi, b_lo, 0);
        cvt8(q4[4], q4[5], b_hi, b_lo, 8);
        c = __builtin_amdgcn_wmma_f32_16x16x32_bf16(false, a_hi[kc], false, b_hi,
                                                    (short)0, c, false, false);
        c = __builtin_amdgcn_wmma_f32_16x16x32_bf16(false, a_hi[kc], false, b_lo,
                                                    (short)0, c, false, false);
        c = __builtin_amdgcn_wmma_f32_16x16x32_bf16(false, a_lo[kc], false, b_hi,
                                                    (short)0, c, false, false);
      }
    }

    // Online rescaled update: track min cost (= max of -cost).
#pragma unroll
    for (int v = 0; v < 8; ++v) {
      float cost = x2r[v] + y2c - 2.0f * c[v];
      float mn   = fminf(m[v], cost);
      float sc   = __expf(mn - m[v]);   // <= 1 (0 on first tile: m=inf, s=w=0)
      float e    = __expf(mn - cost);   // <= 1
      s[v] = s[v] * sc + e;
      w[v] = w[v] * sc + cost * e;
      m[v] = mn;
    }
  }

  // ---- Merge across the 16 columns held by lanes of each half ----
#pragma unroll
  for (int off = 1; off < 16; off <<= 1) {
#pragma unroll
    for (int v = 0; v < 8; ++v) {
      float m2 = __shfl_xor(m[v], off, 32);
      float s2 = __shfl_xor(s[v], off, 32);
      float w2 = __shfl_xor(w[v], off, 32);
      float mn = fminf(m[v], m2);
      float e1 = __expf(mn - m[v]);
      float e2 = __expf(mn - m2);
      s[v] = s[v] * e1 + s2 * e2;
      w[v] = w[v] * e1 + w2 * e2;
      m[v] = mn;
    }
  }

  // ---- Merge the 8 waves (disjoint column sets) via LDS ----
  __shared__ float red_m[NWAVES][16];
  __shared__ float red_s[NWAVES][16];
  __shared__ float red_w[NWAVES][16];
  if (lr == 0) {
#pragma unroll
    for (int v = 0; v < 8; ++v) {
      int row = half * 8 + v;
      red_m[wave][row] = m[v];
      red_s[wave][row] = s[v];
      red_w[wave][row] = w[v];
    }
  }
  __syncthreads();

  if (threadIdx.x < 16) {
    int row = threadIdx.x;
    float mm = red_m[0][row], ss = red_s[0][row], ww = red_w[0][row];
#pragma unroll
    for (int wv = 1; wv < NWAVES; ++wv) {
      float m2 = red_m[wv][row], s2 = red_s[wv][row], w2 = red_w[wv][row];
      float mn = fminf(mm, m2);
      float e1 = __expf(mn - mm);
      float e2 = __expf(mn - m2);
      ss = ss * e1 + s2 * e2;
      ww = ww * e1 + w2 * e2;
      mm = mn;
    }
    float rowval = ww / ss;   // sum_j cost * softmax_j for this row
#pragma unroll
    for (int off = 1; off < 16; off <<= 1) rowval += __shfl_xor(rowval, off, 32);
    if (row == 0) atomicAdd(accum, rowval);
  }
}

__global__ void ct_finalize_kernel(const float* __restrict__ ws,
                                   float* __restrict__ out) {
  if (threadIdx.x == 0) {
    // RHO = 0.5, N = M = 8192
    out[0] = 0.5f * (ws[0] + ws[1]) / (float)NROWS;
  }
}

// ---- host launch ----------------------------------------------------------

extern "C" void kernel_launch(void* const* d_in, const int* in_sizes, int n_in,
                              void* d_out, int out_size, void* d_ws, size_t ws_size,
                              hipStream_t stream) {
  const float* X = (const float*)d_in[0];
  const float* Y = (const float*)d_in[1];
  float* ws  = (float*)d_ws;
  float* out = (float*)d_out;
  float* nx2 = ws + 16;
  float* ny2 = ws + 16 + NROWS;

  ct_norms_kernel<<<2 * NROWS / 8, 256, 0, stream>>>(X, Y, ws);

  const bool split = ws_size >= WS_NEEDED;   // constant per run -> deterministic
  if (split) {
    __bf16* hx = (__bf16*)((char*)d_ws + WS_BF16_OFF);
    __bf16* lx = (__bf16*)((char*)hx + MAT_BYTES);
    __bf16* hy = (__bf16*)((char*)lx + MAT_BYTES);
    __bf16* ly = (__bf16*)((char*)hy + MAT_BYTES);
    ct_split_kernel<<<(2 * NROWS * KDIM / 4) / 256, 256, 0, stream>>>(
        X, Y, hx, lx, hy, ly);
    // forward plan: rows of X, softmax over Y
    ct_main_kernel<true><<<NROWS / 16, 256, 0, stream>>>(
        X, Y, hx, lx, hy, ly, nx2, ny2, ws + 0);
    // backward plan == transposed problem (swap X and Y)
    ct_main_kernel<true><<<NROWS / 16, 256, 0, stream>>>(
        Y, X, hy, ly, hx, lx, ny2, nx2, ws + 1);
  } else {
    ct_main_kernel<false><<<NROWS / 16, 256, 0, stream>>>(
        X, Y, nullptr, nullptr, nullptr, nullptr, nx2, ny2, ws + 0);
    ct_main_kernel<false><<<NROWS / 16, 256, 0, stream>>>(
        Y, X, nullptr, nullptr, nullptr, nullptr, ny2, nx2, ws + 1);
  }
  ct_finalize_kernel<<<1, 32, 0, stream>>>(ws, out);
}
